// Fuse_29137058136121
// MI455X (gfx1250) — matrix operand order repaired and checked by reading, not compiled, not run on hardware
//
#include <hip/hip_runtime.h>
#include <math.h>

#define B_  512
#define NI  36
#define NT  50
#define D_  1024

typedef __attribute__((ext_vector_type(16))) __bf16          v16bf;
typedef __attribute__((ext_vector_type(8)))  float           v8f;
typedef __attribute__((ext_vector_type(8)))  unsigned short  us8;

__device__ __forceinline__ unsigned short f2bf(float f) {
    union { float f; unsigned int u; } c; c.f = f;
    unsigned int r = c.u + 0x7FFFu + ((c.u >> 16) & 1u);   // round-to-nearest-even
    return (unsigned short)(r >> 16);
}
__device__ __forceinline__ v16bf mk16(us8 lo, us8 hi) {
    union { us8 h[2]; v16bf v; } u; u.h[0] = lo; u.h[1] = hi; return u.v;
}

// ---------------------------------------------------------------------------
// bf16 WMMA GEMM: C[M,1024] = A[M,1024] * W[1024,1024] + bias (+ epilogue)
// Block tile 128x64, 8 waves (wave32), each wave 32x32 = 2x2 WMMA 16x16x32 tiles.
// AF32: A is fp32 in global (converted to bf16 while staging to LDS),
//       else A is bf16 (ushort) in global.
// NORM: store bf16 C and atomically accumulate per-row sum(val^2)
// else: C = val + resid, stored fp32.
// ---------------------------------------------------------------------------
template <bool AF32, bool NORM>
__global__ __launch_bounds__(256)
void gemm_bf16_k(const float* __restrict__ Af, const unsigned short* __restrict__ Ah,
                 const float* __restrict__ W,  const float* __restrict__ bias,
                 unsigned short* __restrict__ Cbf, float* __restrict__ sumsq,
                 const float* __restrict__ resid, float* __restrict__ Cout)
{
    constexpr int KT = 32, AS = 40, BS = 40;       // LDS row strides (16B aligned)
    __shared__ unsigned short As[128 * AS];        // [row][k]
    __shared__ unsigned short Bs[64 * BS];         // [n][k] (transposed W tile)

    const int tid  = threadIdx.x;
    const int lane = tid & 31, w = tid >> 5;
    const int lm   = lane & 15, lh = lane >> 4;
    const int wm   = (w & 3) * 32, wn = (w >> 2) * 32;
    const size_t rowBase = (size_t)blockIdx.x * 128;
    const int    colBase = blockIdx.y * 64;

    v8f acc[2][2] = {};

    for (int k0 = 0; k0 < D_; k0 += KT) {
        // ---- light prefetch of the next k-step's tiles (global_prefetch_b8) ----
        if (k0 + KT < D_) {
            if (tid < 128) {
                if constexpr (AF32)
                    __builtin_prefetch(Af + (rowBase + tid) * (size_t)D_ + k0 + KT, 0, 1);
                else
                    __builtin_prefetch(Ah + (rowBase + tid) * (size_t)D_ + k0 + KT, 0, 1);
            } else if (tid < 160) {
                __builtin_prefetch(W + (size_t)(k0 + KT + (tid - 128)) * D_ + colBase, 0, 1);
            }
        }
        // ---- stage A tile (128 x 32) as bf16 ----
        if constexpr (AF32) {
            for (int i = tid; i < 1024; i += 256) {
                int r = i >> 3, c = (i & 7) << 2;
                const float4 f = *(const float4*)(Af + (rowBase + r) * (size_t)D_ + k0 + c);
                unsigned short* p = &As[r * AS + c];
                p[0] = f2bf(f.x); p[1] = f2bf(f.y); p[2] = f2bf(f.z); p[3] = f2bf(f.w);
            }
        } else {
            for (int i = tid; i < 512; i += 256) {
                int r = i >> 2, c = (i & 3) << 3;
                *(us8*)&As[r * AS + c] = *(const us8*)(Ah + (rowBase + r) * (size_t)D_ + k0 + c);
            }
        }
        // ---- stage B tile (32 x 64) fp32 -> bf16, transposed to [n][k] ----
        for (int i = tid; i < 512; i += 256) {
            int kk = i >> 4, c = (i & 15) << 2;
            const float4 f = *(const float4*)(W + (size_t)(k0 + kk) * D_ + colBase + c);
            Bs[(c + 0) * BS + kk] = f2bf(f.x);
            Bs[(c + 1) * BS + kk] = f2bf(f.y);
            Bs[(c + 2) * BS + kk] = f2bf(f.z);
            Bs[(c + 3) * BS + kk] = f2bf(f.w);
        }
        __syncthreads();

        // ---- build fragments per documented wave32 layouts ----
        v16bf a[2], bfr[2];
        for (int ti = 0; ti < 2; ++ti) {           // A: lane = row, k chunks h*8 / 16+h*8
            int ar = wm + ti * 16 + lm;
            a[ti] = mk16(*(const us8*)&As[ar * AS + lh * 8],
                         *(const us8*)&As[ar * AS + 16 + lh * 8]);
        }
        for (int tj = 0; tj < 2; ++tj) {           // B: lane = col, k chunks h*8 / 16+h*8
            int bn = wn + tj * 16 + lm;
            bfr[tj] = mk16(*(const us8*)&Bs[bn * BS + lh * 8],
                           *(const us8*)&Bs[bn * BS + 16 + lh * 8]);
        }
        for (int ti = 0; ti < 2; ++ti)
            for (int tj = 0; tj < 2; ++tj)
                acc[ti][tj] = __builtin_amdgcn_wmma_f32_16x16x32_bf16(
                    false, a[ti], false, bfr[tj], (short)0, acc[ti][tj], false, false);
        __syncthreads();
    }

    // ---- epilogue: C layout lane L -> col lm, VGPR v -> row v + 8*lh ----
    for (int ti = 0; ti < 2; ++ti) {
        for (int v = 0; v < 8; ++v) {
            const size_t gm = rowBase + wm + ti * 16 + v + lh * 8;
            float s2 = 0.f;
            for (int tj = 0; tj < 2; ++tj) {
                const int gn = colBase + wn + tj * 16 + lm;
                float val = acc[ti][tj][v] + bias[gn];
                const size_t idx = gm * (size_t)D_ + gn;
                if constexpr (NORM) {
                    Cbf[idx] = f2bf(val);
                    s2 += val * val;
                } else {
                    Cout[idx] = val + resid[idx];
                }
            }
            if constexpr (NORM) {   // reduce 16 lanes (same row) then one atomic/row
                s2 += __shfl_xor(s2, 1);
                s2 += __shfl_xor(s2, 2);
                s2 += __shfl_xor(s2, 4);
                s2 += __shfl_xor(s2, 8);
                if (lm == 0) atomicAdd(&sumsq[gm], s2);
            }
        }
    }
}

// ---------------------------------------------------------------------------
// Per-batch attention + gating. One block (8 waves) per batch.
//   att = (x . y^T) / max(|x||y|, eps)  via WMMA (3x4 16x16 tiles, K=1024)
//   dual masked softmax, att_x/att_y (K=50/36 fp32), sigmoid gate,
//   writes gated activations as bf16 IN PLACE over x_bf / y_bf.
// Out-of-range tile rows/cols are handled by CLAMPING the source row index
// (no predication -> no exec juggling; garbage lands only in unused attS
// entries: rows >=36 / cols >=50 are never read downstream).
// ---------------------------------------------------------------------------
__global__ __launch_bounds__(256)
void att_fuse_k(const unsigned short* __restrict__ x_bf, const unsigned short* __restrict__ y_bf,
                const float* __restrict__ wxsq, const float* __restrict__ wysq,
                const float* __restrict__ imgE, const float* __restrict__ txtE,
                const unsigned char* __restrict__ x_mask, const unsigned char* __restrict__ y_mask,
                unsigned short* __restrict__ g_x, unsigned short* __restrict__ g_y)
{
    __shared__ float attS[48 * 64];
    __shared__ float att1s[36 * 52];
    __shared__ float att2s[50 * 40];

    const int b = blockIdx.x;
    const int tid = threadIdx.x, lane = tid & 31, w = tid >> 5;
    const int lm = lane & 15, lh = lane >> 4;
    const unsigned short* xb = x_bf + (size_t)b * NI * D_;
    const unsigned short* yb = y_bf + (size_t)b * NT * D_;

    // step 1: raw scores via WMMA (reads straight from L2-resident bf16)
    for (int t = w; t < 12; t += 8) {
        const int ti = t >> 2, tj = t & 3;
        const int an = min(ti * 16 + lm, NI - 1);   // clamp: no predication
        const int bm = min(tj * 16 + lm, NT - 1);
        const unsigned short* ap = xb + (size_t)an * D_ + lh * 8;   // A lo chunk base
        const unsigned short* bp = yb + (size_t)bm * D_ + lh * 8;   // B lo chunk base
        v8f acc = {};
        #pragma unroll 4
        for (int k0 = 0; k0 < D_; k0 += 32) {
            us8 alo = *(const us8*)(ap + k0);
            us8 ahi = *(const us8*)(ap + k0 + 16);
            us8 blo = *(const us8*)(bp + k0);
            us8 bhi = *(const us8*)(bp + k0 + 16);
            acc = __builtin_amdgcn_wmma_f32_16x16x32_bf16(
                false, mk16(alo, ahi), false, mk16(blo, bhi), (short)0, acc, false, false);
        }
        for (int v = 0; v < 8; ++v)
            attS[(ti * 16 + v + lh * 8) * 64 + tj * 16 + lm] = acc[v];
    }
    __syncthreads();

    // step 2: cosine scaling
    for (int i = tid; i < NI * NT; i += 256) {
        int n = i / NT, m = i % NT;
        float den = fmaxf(sqrtf(wxsq[(size_t)b * NI + n]) * sqrtf(wysq[(size_t)b * NT + m]), 1e-8f);
        attS[n * 64 + m] /= den;
    }
    __syncthreads();

    // step 3: masked softmaxes (att1 over txt dim, att2 over img dim)
    if (tid < NI) {
        int n = tid;
        const unsigned char* ym = y_mask + (size_t)b * NT;
        float mx = -3.4e38f;
        for (int m = 0; m < NT; ++m) {
            float a = ym[m] ? -1e9f : attS[n * 64 + m];
            att1s[n * 52 + m] = a; mx = fmaxf(mx, a);
        }
        float s = 0.f;
        for (int m = 0; m < NT; ++m) { float e = expf(att1s[n * 52 + m] - mx); att1s[n * 52 + m] = e; s += e; }
        float inv = 1.f / s;
        for (int m = 0; m < NT; ++m) att1s[n * 52 + m] *= inv;
    }
    if (tid >= 64 && tid < 64 + NT) {
        int m = tid - 64;
        const unsigned char* xm = x_mask + (size_t)b * NI;
        float mx = -3.4e38f;
        for (int n = 0; n < NI; ++n) {
            float a = xm[n] ? -1e9f : attS[n * 64 + m];
            att2s[m * 40 + n] = a; mx = fmaxf(mx, a);
        }
        float s = 0.f;
        for (int n = 0; n < NI; ++n) { float e = expf(att2s[m * 40 + n] - mx); att2s[m * 40 + n] = e; s += e; }
        float inv = 1.f / s;
        for (int n = 0; n < NI; ++n) att2s[m * 40 + n] *= inv;
    }
    __syncthreads();

    // step 4: att_x = att1 @ txtE, gate, write bf16 (overwrites x_bf for this batch)
    for (int i = tid; i < NI * D_; i += 256) {
        int n = i >> 10, d = i & 1023;
        const float* tE = txtE + (size_t)b * NT * D_ + d;
        float s = 0.f;
        for (int m = 0; m < NT; ++m) s += att1s[n * 52 + m] * tE[(size_t)m * D_];
        float ie = imgE[((size_t)b * NI + n) * D_ + d];
        float gate = 1.f / (1.f + expf(-(s * ie)));
        g_x[((size_t)b * NI + n) * D_ + d] = f2bf(gate * s);
    }
    // step 5: att_y = att2 @ imgE, gate, write bf16 (overwrites y_bf for this batch)
    for (int i = tid; i < NT * D_; i += 256) {
        int m = i >> 10, d = i & 1023;
        const float* iE = imgE + (size_t)b * NI * D_ + d;
        float s = 0.f;
        for (int n = 0; n < NI; ++n) s += att2s[m * 40 + n] * iE[(size_t)n * D_];
        float te = txtE[((size_t)b * NT + m) * D_ + d];
        float gate = 1.f / (1.f + expf(-(s * te)));
        g_y[((size_t)b * NT + m) * D_ + d] = f2bf(gate * s);
    }
}

__global__ void zero_k(float* p, int n) {
    int i = blockIdx.x * 256 + threadIdx.x;
    if (i < n) p[i] = 0.f;
}

extern "C" void kernel_launch(void* const* d_in, const int* in_sizes, int n_in,
                              void* d_out, int out_size, void* d_ws, size_t ws_size,
                              hipStream_t stream)
{
    (void)in_sizes; (void)n_in; (void)out_size; (void)ws_size;
    const float*         imgE   = (const float*)d_in[0];
    const float*         txtE   = (const float*)d_in[1];
    const unsigned char* x_mask = (const unsigned char*)d_in[2];
    const unsigned char* y_mask = (const unsigned char*)d_in[3];
    const float* W_img  = (const float*)d_in[4];
    const float* b_img  = (const float*)d_in[5];
    const float* W_txt  = (const float*)d_in[6];
    const float* b_txt  = (const float*)d_in[7];
    const float* W_img1 = (const float*)d_in[8];
    const float* b_img1 = (const float*)d_in[9];
    const float* W_txt1 = (const float*)d_in[10];
    const float* b_txt1 = (const float*)d_in[11];

    const size_t MI = (size_t)B_ * NI;   // 18432 rows
    const size_t MT = (size_t)B_ * NT;   // 25600 rows

    // workspace layout (~90.4 MB): x_bf|g_x, y_bf|g_y (aliased), wx^2, wy^2
    char* ws = (char*)d_ws;
    unsigned short* x_bf = (unsigned short*)ws;
    unsigned short* y_bf = (unsigned short*)(ws + MI * D_ * 2);
    float* wxsq = (float*)(ws + (MI + MT) * D_ * 2);
    float* wysq = wxsq + MI;

    float* out_img = (float*)d_out;
    float* out_txt = out_img + MI * D_;

    zero_k<<<dim3((unsigned)((MI + MT + 255) / 256)), 256, 0, stream>>>(wxsq, (int)(MI + MT));

    gemm_bf16_k<true, true><<<dim3((unsigned)(MI / 128), D_ / 64), 256, 0, stream>>>(
        imgE, nullptr, W_img, b_img, x_bf, wxsq, nullptr, nullptr);
    gemm_bf16_k<true, true><<<dim3((unsigned)(MT / 128), D_ / 64), 256, 0, stream>>>(
        txtE, nullptr, W_txt, b_txt, y_bf, wysq, nullptr, nullptr);

    att_fuse_k<<<dim3(B_), 256, 0, stream>>>(x_bf, y_bf, wxsq, wysq, imgE, txtE,
                                             x_mask, y_mask, x_bf, y_bf);

    gemm_bf16_k<false, false><<<dim3((unsigned)(MI / 128), D_ / 64), 256, 0, stream>>>(
        nullptr, x_bf, W_img1, b_img1, nullptr, nullptr, imgE, out_img);
    gemm_bf16_k<false, false><<<dim3((unsigned)(MT / 128), D_ / 64), 256, 0, stream>>>(
        nullptr, y_bf, W_txt1, b_txt1, nullptr, nullptr, txtE, out_txt);
}
